// GravNet_30666066493850
// MI455X (gfx1250) — matrix operand order repaired and checked by reading, not compiled, not run on hardware
//
#include <hip/hip_runtime.h>
#include <hip/hip_bf16.h>
#include <math.h>

typedef __attribute__((ext_vector_type(16))) __bf16 v16bf;
typedef __attribute__((ext_vector_type(8)))  __bf16 v8bf;
typedef __attribute__((ext_vector_type(8)))  float  v8f;

#define NROWS  32768   // B*P
#define NGRAPH 128
#define NPTS   256
#define KNN    20

__device__ __forceinline__ v8bf zero_v8bf() {
  v8bf z;
#pragma unroll
  for (int e = 0; e < 8; ++e) z[e] = (__bf16)0.0f;
  return z;
}

// CDNA5 async copy: global -> LDS, 16 bytes per lane, tracked by ASYNCcnt.
__device__ __forceinline__ void async_b128_to_lds(const void* gsrc, void* ldst) {
  unsigned lds = (unsigned)(unsigned long long)ldst;  // low 32b of flat shared addr = LDS offset
  asm volatile("global_load_async_to_lds_b128 %0, %1, off"
               :: "v"(lds), "v"((unsigned long long)gsrc)
               : "memory");
}
__device__ __forceinline__ void wait_asynccnt0() {
  asm volatile("s_wait_asynccnt 0x0" ::: "memory");
}

// ---------------- conversion / packing kernels ----------------
__global__ void f32_to_bf16_kernel(const float* __restrict__ src, __bf16* __restrict__ dst, int n) {
  int i = blockIdx.x * 256 + threadIdx.x;
  if (i < n) dst[i] = (__bf16)src[i];
}

// Pack Ws [Fin,4] and Wh [Fin,64] into a single [Fin,80] bf16 matrix (+bias[80]).
__global__ void pack_sh_kernel(const float* __restrict__ Ws, const float* __restrict__ bs,
                               const float* __restrict__ Wh, const float* __restrict__ bh,
                               __bf16* __restrict__ W, float* __restrict__ bias, int Fin) {
  int i = blockIdx.x * 256 + threadIdx.x;
  if (i < Fin * 80) {
    int r = i / 80, c = i % 80;
    float v = 0.f;
    if (c < 4)       v = Ws[r * 4 + c];
    else if (c < 68) v = Wh[r * 64 + (c - 4)];
    W[i] = (__bf16)v;
  }
  if (i < 80) {
    float v = 0.f;
    if (i < 4)       v = bs[i];
    else if (i < 68) v = bh[i - 4];
    bias[i] = v;
  }
}

// ---------------- tiled bf16 WMMA GEMM (async-staged, double-buffered) ----------------
// C[M,N] = op( concat_k(A0[:, :K0], A1[:, :Ktot-K0]) @ Bw[Ktot,N] + bias )
// Block tile 128(M) x 128(N), k-step 32, 8 waves (4 row-groups of 32 x 2 col-groups of 64).
// Each wave: 2 A-frags x 4 B-frags = 8 v_wmma_f32_16x16x32_bf16 per k-step.
// A/B tiles staged via global_load_async_to_lds_b128 (ASYNCcnt), double buffered.
// SPLIT: A fetched from A0 (k<K0) / A1 (k>=K0); K0 % 8 == 0 so 16B chunks never straddle.
// GUARD: bounds checks for Ktot % 32 != 0 or N % 128 != 0; hot GEMMs compile them out.
template <bool OUT_BF16, bool RELU, bool ACCUM, bool SPLIT, bool GUARD>
__global__ __launch_bounds__(256)
void gemm_bf16_wmma(const __bf16* __restrict__ A0, int lda0, int K0,
                    const __bf16* __restrict__ A1, int lda1,
                    const __bf16* __restrict__ Bw, int ldb,
                    const float* __restrict__ bias,
                    void* __restrict__ Cp, int ldc,
                    int N, int Ktot) {
  __shared__ __bf16 lA[2][128][40];
  __shared__ __bf16 lB[2][32][136];
  const int tid  = threadIdx.x;
  const int lane = tid & 31;
  const int wave = tid >> 5;
  const int wm   = wave & 3;     // row group (32 rows)
  const int wn   = wave >> 2;    // col group (64 cols)
  const int hlf  = lane >> 4;
  const int m16  = lane & 15;
  const int rowBase = blockIdx.y * 128;
  const int colBase = blockIdx.x * 128;

  v8f acc[2][4];
#pragma unroll
  for (int f = 0; f < 2; ++f)
#pragma unroll
    for (int t = 0; t < 4; ++t)
#pragma unroll
      for (int r = 0; r < 8; ++r) acc[f][t][r] = 0.f;

  if (ACCUM) {
    const float* Cf = (const float*)Cp;
#pragma unroll
    for (int f = 0; f < 2; ++f)
#pragma unroll
      for (int t = 0; t < 4; ++t) {
        int gcol = colBase + wn * 64 + t * 16 + m16;
#pragma unroll
        for (int r = 0; r < 8; ++r) {
          int grow = rowBase + wm * 32 + f * 16 + hlf * 8 + r;
          acc[f][t][r] = Cf[(size_t)grow * ldc + gcol];
        }
      }
  }

  // A stage: 128 rows x 4 chunks of 8 elems (16B); 2 chunks per thread
  const int rowA  = tid >> 1;
  const int colA0 = (tid & 1) * 16;
  // B stage: 32 krows x 8 chunks of 16 elems (32B); 1 per thread
  const int bk = tid >> 3, bc = tid & 7;
  const int bcol = colBase + bc * 16;
  const bool bColOK = !GUARD || (bcol < N);   // N % 16 == 0 -> whole-chunk predicate

  const __bf16* arow0 = A0 + (size_t)(rowBase + rowA) * lda0;
  const __bf16* arow1 = A1 + (size_t)(rowBase + rowA) * lda1;

  auto stageA = [&](int k0, int s) {
#pragma unroll
    for (int h = 0; h < 2; ++h) {
      int colc = colA0 + h * 8;
      int kk0  = k0 + colc;                  // K0 % 8 == 0 -> never straddles A0/A1
      if (!GUARD || kk0 < Ktot) {
        const __bf16* src;
        if (SPLIT) src = (kk0 < K0) ? (arow0 + kk0) : (arow1 + (kk0 - K0));
        else       src = arow0 + kk0;
        async_b128_to_lds(src, &lA[s][rowA][colc]);
      } else {
        *(v8bf*)&lA[s][rowA][colc] = zero_v8bf();
      }
    }
  };
  auto stageB = [&](int k0, int s) {
    int gk = k0 + bk;
    if ((!GUARD || gk < Ktot) && bColOK) {
      const __bf16* src = Bw + (size_t)gk * ldb + bcol;
      async_b128_to_lds(src,     &lB[s][bk][bc * 16]);
      async_b128_to_lds(src + 8, &lB[s][bk][bc * 16 + 8]);
    } else {
      *(v8bf*)&lB[s][bk][bc * 16]     = zero_v8bf();
      *(v8bf*)&lB[s][bk][bc * 16 + 8] = zero_v8bf();
    }
  };

  stageA(0, 0);
  stageB(0, 0);
  wait_asynccnt0();
  __syncthreads();

  int cur = 0;
  for (int k0 = 0; k0 < Ktot; k0 += 32) {
    const int nxt = cur ^ 1;
    if (k0 + 32 < Ktot) {        // prefetch next tile while computing this one
      stageA(k0 + 32, nxt);
      stageB(k0 + 32, nxt);
    }

    // A fragments per ISA 16-bit 16x32 layout
    v16bf af[2];
#pragma unroll
    for (int f = 0; f < 2; ++f)
#pragma unroll
      for (int i = 0; i < 8; ++i) {
        int kb = ((i < 4) ? (2 * i) : (16 + 2 * (i - 4))) + hlf * 8;
        af[f][2 * i]     = lA[cur][wm * 32 + f * 16 + m16][kb];
        af[f][2 * i + 1] = lA[cur][wm * 32 + f * 16 + m16][kb + 1];
      }
    const int kB = hlf * 8;  // B: K pairs over VGPRs (same order as A), N over lanes
#pragma unroll
    for (int t = 0; t < 4; ++t) {
      v16bf bfv;
      const int c0 = wn * 64 + t * 16;
#pragma unroll
      for (int j = 0; j < 8; ++j) {
        int kb = ((j < 4) ? (2 * j) : (16 + 2 * (j - 4))) + kB;
        bfv[2 * j]     = lB[cur][kb][c0 + m16];
        bfv[2 * j + 1] = lB[cur][kb + 1][c0 + m16];
      }
#pragma unroll
      for (int f = 0; f < 2; ++f)
        acc[f][t] = __builtin_amdgcn_wmma_f32_16x16x32_bf16(false, af[f], false, bfv,
                                                            (short)0, acc[f][t], false, false);
    }

    wait_asynccnt0();            // prefetched tile resident in LDS
    __syncthreads();             // everyone done reading lX[cur]
    cur = nxt;
  }

#pragma unroll
  for (int f = 0; f < 2; ++f)
#pragma unroll
    for (int t = 0; t < 4; ++t) {
      int gcol = colBase + wn * 64 + t * 16 + m16;
      if (!GUARD || gcol < N) {
        float bv = bias ? bias[gcol] : 0.f;
#pragma unroll
        for (int r = 0; r < 8; ++r) {
          int grow = rowBase + wm * 32 + f * 16 + hlf * 8 + r;
          float v = acc[f][t][r] + bv;
          if (RELU) v = fmaxf(v, 0.f);
          if (OUT_BF16) ((__bf16*)Cp)[(size_t)grow * ldc + gcol] = (__bf16)v;
          else          ((float*)Cp)[(size_t)grow * ldc + gcol] = v;
        }
      }
    }
}

// ---------------- kNN (one block per graph, thread per point) ----------------
__global__ __launch_bounds__(256)
void knn_kernel(const float* __restrict__ sh, int* __restrict__ nidx, float* __restrict__ nwgt) {
  __shared__ float ls[NPTS][4];
  const int g = blockIdx.x;
  const int p = threadIdx.x;
  const float* row = sh + ((size_t)g * NPTS + p) * 80;
  float s0 = row[0], s1 = row[1], s2 = row[2], s3 = row[3];
  ls[p][0] = s0; ls[p][1] = s1; ls[p][2] = s2; ls[p][3] = s3;
  __syncthreads();
  float nd[KNN]; int ni[KNN];
#pragma unroll
  for (int j = 0; j < KNN; ++j) { nd[j] = 3.4e38f; ni[j] = 0; }
  for (int q = 0; q < NPTS; ++q) {
    float dx = s0 - ls[q][0];
    float dy = s1 - ls[q][1];
    float dz = s2 - ls[q][2];
    float dw = s3 - ls[q][3];
    float d2 = dx*dx + dy*dy + dz*dz + dw*dw;
    if (d2 < nd[KNN-1]) {               // register-resident unrolled insertion
      nd[KNN-1] = d2; ni[KNN-1] = q;
#pragma unroll
      for (int j = KNN-1; j > 0; --j) {
        if (nd[j] < nd[j-1]) {
          float td = nd[j]; nd[j] = nd[j-1]; nd[j-1] = td;
          int   ti = ni[j]; ni[j] = ni[j-1]; ni[j-1] = ti;
        }
      }
    }
  }
  size_t base = ((size_t)g * NPTS + p) * KNN;
#pragma unroll
  for (int j = 0; j < KNN; ++j) {
    nidx[base + j] = ni[j];
    float d = nd[j] > 0.f ? nd[j] : 0.f;
    nwgt[base + j] = __expf(-10.0f * d);
  }
}

// ---------------- neighbor aggregation (weighted mean + max) ----------------
__global__ __launch_bounds__(256)
void aggregate_kernel(const float* __restrict__ sh, const int* __restrict__ nidx,
                      const float* __restrict__ nwgt, __bf16* __restrict__ agg) {
  __shared__ __bf16 lh[NPTS][66];   // pad 2 -> conflict-free gathers
  const int g = blockIdx.x;
  const int p = threadIdx.x;
  const float* hr = sh + ((size_t)g * NPTS + p) * 80 + 4;
#pragma unroll 8
  for (int c = 0; c < 64; ++c) lh[p][c] = (__bf16)hr[c];
  __syncthreads();
  size_t base = ((size_t)g * NPTS + p) * KNN;
  int nb[KNN]; float wv[KNN];
#pragma unroll
  for (int j = 0; j < KNN; ++j) { nb[j] = nidx[base + j]; wv[j] = nwgt[base + j]; }
  __bf16* out = agg + ((size_t)g * NPTS + p) * 128;
  for (int cg = 0; cg < 4; ++cg) {
    float mn[16], mx[16];
#pragma unroll
    for (int e = 0; e < 16; ++e) { mn[e] = 0.f; mx[e] = -3.4e38f; }
    for (int j = 0; j < KNN; ++j) {
      float w = wv[j];
      const __bf16* hn = &lh[nb[j]][cg * 16];
#pragma unroll
      for (int e = 0; e < 16; ++e) {
        float v = (float)hn[e] * w;
        mn[e] += v;
        mx[e] = fmaxf(mx[e], v);
      }
    }
#pragma unroll
    for (int e = 0; e < 16; ++e) {
      out[cg * 16 + e]      = (__bf16)(mn[e] * (1.0f / KNN));
      out[64 + cg * 16 + e] = (__bf16)mx[e];
    }
  }
}

// ---------------- multi-pool: min/max/mean/sum/std/median ----------------
__device__ __forceinline__ unsigned mapf(float f) {
  unsigned b = __float_as_uint(f);
  return (b & 0x80000000u) ? ~b : (b | 0x80000000u);
}
__device__ __forceinline__ float unmapf(unsigned u) {
  unsigned b = (u & 0x80000000u) ? (u ^ 0x80000000u) : ~u;
  return __uint_as_float(b);
}

__global__ __launch_bounds__(64)
void pool_kernel(const float* __restrict__ mid, __bf16* __restrict__ pooled) {
  __shared__ unsigned lv[NPTS][64];       // order-preserving keys, 64KB
  const int g  = blockIdx.y;
  const int cg = blockIdx.x;              // 8 groups of 64 channels
  const int c  = threadIdx.x;
  const int ch = cg * 64 + c;
  float mnv = 3.4e38f, mxv = -3.4e38f, sum = 0.f, sumsq = 0.f;
  for (int p = 0; p < NPTS; ++p) {
    float v = mid[((size_t)g * NPTS + p) * 512 + ch];
    lv[p][c] = mapf(v);
    mnv = fminf(mnv, v);
    mxv = fmaxf(mxv, v);
    sum += v;
    sumsq += v * v;
  }
  __syncthreads();
  // 128th-smallest (0-based idx 127, 'lower') via bitwise binary search
  unsigned lo = 0u, hi = 0xFFFFFFFFu;
  for (int it = 0; it < 32; ++it) {
    unsigned mv = lo + ((hi - lo) >> 1);
    int cnt = 0;
    for (int p = 0; p < NPTS; ++p) cnt += (lv[p][c] <= mv) ? 1 : 0;
    if (cnt >= 128) hi = mv; else lo = mv + 1;
  }
  float med  = unmapf(hi);
  float mean = sum * (1.0f / NPTS);
  float var  = sumsq * (1.0f / NPTS) - mean * mean;
  float sd   = sqrtf(fmaxf(var, 0.f) + 1e-5f);
  __bf16* out = pooled + (size_t)g * 3072;
  out[0 * 512 + ch] = (__bf16)mnv;
  out[1 * 512 + ch] = (__bf16)mxv;
  out[2 * 512 + ch] = (__bf16)mean;
  out[3 * 512 + ch] = (__bf16)sum;
  out[4 * 512 + ch] = (__bf16)sd;
  out[5 * 512 + ch] = (__bf16)med;
}

// ---------------- host orchestration ----------------
extern "C" void kernel_launch(void* const* d_in, const int* in_sizes, int n_in,
                              void* d_out, int out_size, void* d_ws, size_t ws_size,
                              hipStream_t stream) {
  (void)in_sizes; (void)n_in; (void)out_size; (void)ws_size;
  static const int FIN[8]  = {8, 256, 256, 512, 512, 512, 1024, 1024};
  static const int FO[8]   = {256, 256, 512, 512, 512, 1024, 1024, 1024};
  static const int MOFF[9] = {0, 8, 264, 520, 1032, 1544, 2056, 3080, 4104};

  const float* x_f    = (const float*)d_in[0];
  const float* W_mid  = (const float*)d_in[50];
  const float* b_mid  = (const float*)d_in[51];
  const float* W_head = (const float*)d_in[52];
  const float* b_head = (const float*)d_in[53];
  float* out = (float*)d_out;

  char* ws = (char*)d_ws;
  size_t off = 0;
  auto alloc = [&](size_t bytes) { size_t o = off; off += (bytes + 255) & ~(size_t)255; return o; };
  size_t oXA   = alloc((size_t)NROWS * 1024 * 2);
  size_t oXB   = alloc((size_t)NROWS * 1024 * 2);
  size_t oSH   = alloc((size_t)NROWS * 80 * 4);
  size_t oAG   = alloc((size_t)NROWS * 128 * 2);
  size_t oIX   = alloc((size_t)NROWS * KNN * 4);
  size_t oWG   = alloc((size_t)NROWS * KNN * 4);
  size_t oMID  = alloc((size_t)NROWS * 512 * 4);
  size_t oPOOL = alloc((size_t)NGRAPH * 3072 * 2);
  size_t oWSH  = alloc((size_t)4104 * 80 * 2);
  size_t oBSH  = alloc((size_t)8 * 80 * 4);
  size_t oWO   = alloc((size_t)3999744 * 2);
  size_t oWM   = alloc((size_t)5128 * 512 * 2);
  size_t oWH   = alloc((size_t)3072 * 128 * 2);

  __bf16* xA      = (__bf16*)(ws + oXA);
  __bf16* xB      = (__bf16*)(ws + oXB);
  float*  sh      = (float*)(ws + oSH);
  __bf16* agg     = (__bf16*)(ws + oAG);
  int*    nidx    = (int*)(ws + oIX);
  float*  nwgt    = (float*)(ws + oWG);
  float*  mid     = (float*)(ws + oMID);
  __bf16* pooled  = (__bf16*)(ws + oPOOL);
  __bf16* WshAll  = (__bf16*)(ws + oWSH);
  float*  biasSh  = (float*)(ws + oBSH);
  __bf16* WoAll   = (__bf16*)(ws + oWO);
  __bf16* WmidB   = (__bf16*)(ws + oWM);
  __bf16* WheadB  = (__bf16*)(ws + oWH);

  size_t wshOff[8], woOff[8];
  { size_t a = 0, b = 0;
    for (int i = 0; i < 8; ++i) {
      wshOff[i] = a; a += (size_t)FIN[i] * 80;
      woOff[i]  = b; b += (size_t)(FIN[i] + 128) * FO[i];
    } }

  // ---- weight + input conversion ----
  { int n = NROWS * 8;
    f32_to_bf16_kernel<<<(n + 255) / 256, 256, 0, stream>>>(x_f, xA, n); }
  for (int i = 0; i < 8; ++i) {
    const float* Ws = (const float*)d_in[2 + 6 * i];
    const float* bs = (const float*)d_in[3 + 6 * i];
    const float* Wh = (const float*)d_in[4 + 6 * i];
    const float* bh = (const float*)d_in[5 + 6 * i];
    const float* Wo = (const float*)d_in[6 + 6 * i];
    int nsh = FIN[i] * 80;
    pack_sh_kernel<<<(nsh + 255) / 256, 256, 0, stream>>>(Ws, bs, Wh, bh,
        WshAll + wshOff[i], biasSh + i * 80, FIN[i]);
    int nwo = (FIN[i] + 128) * FO[i];
    f32_to_bf16_kernel<<<(nwo + 255) / 256, 256, 0, stream>>>(Wo, WoAll + woOff[i], nwo);
  }
  { int n = 5128 * 512; f32_to_bf16_kernel<<<(n + 255) / 256, 256, 0, stream>>>(W_mid, WmidB, n); }
  { int n = 3072 * 128; f32_to_bf16_kernel<<<(n + 255) / 256, 256, 0, stream>>>(W_head, WheadB, n); }

  // ---- mid = b_mid + x0 @ W_mid[0:8]  (incremental mid; GUARD handles K=8) ----
  {
    dim3 grid(4, NROWS / 128);
    gemm_bf16_wmma<false, false, false, false, true><<<grid, 256, 0, stream>>>(
        xA, 8, 8, xA, 8, WmidB + (size_t)MOFF[0] * 512, 512, b_mid, mid, 512, 512, 8);
  }

  __bf16* xcur = xA;
  __bf16* xnext = xB;
  for (int i = 0; i < 8; ++i) {
    int fi = FIN[i], fo = FO[i];
    { // [s|h] = x @ [Ws|Wh] + [bs|bh]   (N=80 -> GUARD, f32 out)
      dim3 grid(1, NROWS / 128);
      gemm_bf16_wmma<false, false, false, false, true><<<grid, 256, 0, stream>>>(
          xcur, fi, fi, xcur, fi, WshAll + wshOff[i], 80, biasSh + i * 80, sh, 80, 80, fi);
    }
    knn_kernel<<<NGRAPH, NPTS, 0, stream>>>(sh, nidx, nwgt);
    aggregate_kernel<<<NGRAPH, NPTS, 0, stream>>>(sh, nidx, nwgt, agg);
    { // x_next = relu(concat(x, agg) @ Wo + bo)   (SPLIT-A, no guard, bf16 out)
      const float* bo = (const float*)d_in[7 + 6 * i];
      dim3 grid(fo / 128, NROWS / 128);
      gemm_bf16_wmma<true, true, false, true, false><<<grid, 256, 0, stream>>>(
          xcur, fi, fi, agg, 128, WoAll + woOff[i], fo, bo, xnext, fo, fo, fi + 128);
    }
    { // mid += x_next @ W_mid[off_i : off_i+fo]   (ACCUM, no guard)
      dim3 grid(4, NROWS / 128);
      gemm_bf16_wmma<false, false, true, false, false><<<grid, 256, 0, stream>>>(
          xnext, fo, fo, xnext, fo, WmidB + (size_t)MOFF[i + 1] * 512, 512, nullptr,
          mid, 512, 512, fo);
    }
    __bf16* t = xcur; xcur = xnext; xnext = t;
  }

  // ---- pooling then head ----
  {
    dim3 grid(8, NGRAPH);
    pool_kernel<<<grid, 64, 0, stream>>>(mid, pooled);
  }
  { // head: [128,3072] @ [3072,128] + bias -> d_out (f32), single 128x128 block
    dim3 grid(1, 1);
    gemm_bf16_wmma<false, false, false, false, false><<<grid, 256, 0, stream>>>(
        pooled, 3072, 3072, pooled, 3072, WheadB, 128, b_head, out, 128, 128, 3072);
  }
}
